// Decoder_40578851013223
// MI455X (gfx1250) — matrix operand order repaired and checked by reading, not compiled, not run on hardware
//
#include <hip/hip_runtime.h>
#include <math.h>

// Problem constants (match reference)
#define H      768
#define SEQ    512
#define VOC    30522
#define BB     8
#define EXT    (VOC + SEQ)      // 31034
#define COMB   (2 * VOC + SEQ)  // 61556
#define MAXLEN 32
#define NEGV   (-1.0e6f)
#define GENP   30528            // padded N (1908 * 16)
#define NT     1908             // N tiles of 16 (ceil(30522/16))
#define KC     24               // K chunks of 32 (768/32)

typedef __attribute__((ext_vector_type(8)))  float  v8f;
typedef __attribute__((ext_vector_type(16))) __bf16 v16bf;

__device__ __forceinline__ unsigned short f2bf(float f) {
  unsigned int u = __float_as_uint(f);
  u += 0x7FFFu + ((u >> 16) & 1u);   // round-to-nearest-even
  return (unsigned short)(u >> 16);
}

// ---------------------------------------------------------------------------
// One-time prep: pack out_w (V x H fp32) into bf16 WMMA B fragments.
// B matrix is 32x16 (KxN) per chunk: lane l -> N = l%16; hi = l/16;
// VGPR0..3 hold K = hi*8 + 0..7 (pairs), VGPR4..7 hold K = 16 + hi*8 + 0..7.
// Fragment id = (tile*KC + chunk)*32 + lane, 32 bytes each.
// ---------------------------------------------------------------------------
__global__ void k_prep_bpack(const float* __restrict__ out_w,
                             unsigned char* __restrict__ bpack) {
  int id = blockIdx.x * blockDim.x + threadIdx.x;   // exactly NT*KC*32 threads
  int l  = id & 31;
  int c  = (id >> 5) % KC;
  int t  = id / (KC * 32);
  int n  = t * 16 + (l & 15);
  int hi = l >> 4;
  unsigned short* dst = (unsigned short*)(bpack + (size_t)id * 32);
#pragma unroll
  for (int j = 0; j < 16; ++j) {
    int   k = c * 32 + ((j < 8) ? (hi * 8 + j) : (16 + hi * 8 + (j - 8)));
    float v = (n < VOC) ? out_w[(size_t)n * H + k] : 0.0f;
    dst[j] = f2bf(v);
  }
}

// Per-step: pack hidden (8 x 768 fp32, zero-padded to M=16) into bf16 A frags.
// A matrix 16x32 (MxK): lane l -> M = l%16; hi = l/16;
// VGPR0..3 hold K = hi*8 + 0..7 (pairs), VGPR4..7 hold K = 16 + hi*8 + 0..7.
__global__ void k_pack_a(const float* __restrict__ hid,
                         unsigned char* __restrict__ apack) {
  int id = blockIdx.x * blockDim.x + threadIdx.x;   // KC*32 = 768 threads
  if (id >= KC * 32) return;
  int l = id & 31, c = id >> 5;
  int m = l & 15,  hi = l >> 4;
  unsigned short* dst = (unsigned short*)(apack + (size_t)id * 32);
#pragma unroll
  for (int j = 0; j < 16; ++j) {
    int   k = c * 32 + ((j < 8) ? (hi * 8 + j) : (16 + hi * 8 + (j - 8)));
    float v = (m < BB) ? hid[m * H + k] : 0.0f;
    dst[j] = f2bf(v);
  }
}

// ---------------------------------------------------------------------------
// gen = h @ out_w.T + out_b  via v_wmma_f32_16x16x32_bf16.
// One wave per 16-wide N tile, 24 K chunks. D layout: VGPR r, lanes 0-15 ->
// (M=r, N=lane); lanes 16-31 hold the zero-padded M=8..15 rows (discarded).
// ---------------------------------------------------------------------------
__global__ void __launch_bounds__(256)
k_gen(const unsigned char* __restrict__ apack,
      const unsigned char* __restrict__ bpack,
      const float* __restrict__ out_b,
      float* __restrict__ gen) {
  int lane = threadIdx.x & 31;
  int wave = (blockIdx.x * blockDim.x + threadIdx.x) >> 5;
  if (wave >= NT) return;                  // wave-uniform exit
  const int t = wave;
  v8f acc = {0.f, 0.f, 0.f, 0.f, 0.f, 0.f, 0.f, 0.f};
  for (int c = 0; c < KC; ++c) {
    v16bf a = *(const v16bf*)(apack + (((size_t)(c * 32 + lane)) << 5));
    v16bf b = *(const v16bf*)(bpack + (((size_t)((t * KC + c) * 32 + lane)) << 5));
    if (c + 1 < KC)
      __builtin_prefetch(bpack + (((size_t)((t * KC + c + 1) * 32 + lane)) << 5), 0, 1);
    acc = __builtin_amdgcn_wmma_f32_16x16x32_bf16(false, a, false, b,
                                                  (short)0, acc, false, false);
  }
  if (lane < 16) {
    int n = t * 16 + lane;
    if (n < VOC) {
      float bias = out_b[n];
#pragma unroll
      for (int r = 0; r < 8; ++r)
        gen[(size_t)r * GENP + n] = (n == 0) ? NEGV : (acc[r] + bias);
    }
  }
}

// ---------------------------------------------------------------------------
// Attention: th = h@Wa.T + ba ; scores = enc . th ; softmax ; context.
// One block (768 threads) per batch row.
// ---------------------------------------------------------------------------
__global__ void __launch_bounds__(768)
k_attn(const float* __restrict__ enc, const float* __restrict__ w,
       const float* __restrict__ bias, const float* __restrict__ hid,
       float* __restrict__ ctx) {
  __shared__ float th[H];
  __shared__ float sc[SEQ];
  __shared__ float part[64];
  __shared__ float sM, sZ;
  const int b = blockIdx.x, tid = threadIdx.x;
  const float* hb = hid + b * H;
  {
    float a = 0.f;
    const float* wr = w + (size_t)tid * H;
    for (int k = 0; k < H; ++k) a += hb[k] * wr[k];
    th[tid] = a + bias[tid];
  }
  __syncthreads();
  if (tid < SEQ) {
    const float* er = enc + ((size_t)b * SEQ + tid) * H;
    float a = 0.f;
    for (int k = 0; k < H; ++k) a += er[k] * th[k];
    sc[tid] = a;
  }
  __syncthreads();
  if (tid < 64) { float m = -3.4e38f; for (int i = 0; i < 8; ++i) m = fmaxf(m, sc[tid * 8 + i]); part[tid] = m; }
  __syncthreads();
  if (tid == 0) { float m = -3.4e38f; for (int i = 0; i < 64; ++i) m = fmaxf(m, part[i]); sM = m; }
  __syncthreads();
  if (tid < SEQ) sc[tid] = expf(sc[tid] - sM);
  __syncthreads();
  if (tid < 64) { float a = 0.f; for (int i = 0; i < 8; ++i) a += sc[tid * 8 + i]; part[tid] = a; }
  __syncthreads();
  if (tid == 0) { float a = 0.f; for (int i = 0; i < 64; ++i) a += part[i]; sZ = a; }
  __syncthreads();
  {
    float a = 0.f;
    for (int s = 0; s < SEQ; ++s) a += sc[s] * enc[((size_t)b * SEQ + s) * H + tid];
    ctx[b * H + tid] = a / sZ;
  }
}

// Copy scores: th2 = h@Wc.T + bc ; css[b,s] = enc[b,s] . th2
__global__ void __launch_bounds__(768)
k_copy(const float* __restrict__ enc, const float* __restrict__ w,
       const float* __restrict__ bias, const float* __restrict__ hid,
       float* __restrict__ css) {
  __shared__ float th[H];
  const int b = blockIdx.x, tid = threadIdx.x;
  const float* hb = hid + b * H;
  {
    float a = 0.f;
    const float* wr = w + (size_t)tid * H;
    for (int k = 0; k < H; ++k) a += hb[k] * wr[k];
    th[tid] = a + bias[tid];
  }
  __syncthreads();
  if (tid < SEQ) {
    const float* er = enc + ((size_t)b * SEQ + tid) * H;
    float a = 0.f;
    for (int k = 0; k < H; ++k) a += er[k] * th[k];
    css[b * SEQ + tid] = a;
  }
}

// x = [context, sel, emb[idx_c]]
__global__ void k_build_x(const float* __restrict__ ctx, const float* __restrict__ sel,
                          const float* __restrict__ emb, const int* __restrict__ smp,
                          float* __restrict__ x) {
  int id = blockIdx.x * blockDim.x + threadIdx.x;
  if (id >= BB * 3 * H) return;
  int b = id / (3 * H), j = id % (3 * H);
  float v;
  if (j < H)            v = ctx[b * H + j];
  else if (j < 2 * H)   v = sel[b * H + (j - H)];
  else {
    int idx = smp[b];
    int ic  = (idx > VOC) ? 3 : idx;     // reference: strict >
    if (ic >= VOC) ic = VOC - 1;         // JAX clamp for idx == V
    if (ic < 0) ic = 0;
    v = emb[(size_t)ic * H + (j - 2 * H)];
  }
  x[b * 3 * H + j] = v;
}

// GRU matmuls: gi = x@Wih.T + bih ; gh = h@Whh.T + bhh (thread per gate row g)
__global__ void __launch_bounds__(256)
k_gru_mm(const float* __restrict__ x, const float* __restrict__ hid,
         const float* __restrict__ w_ih, const float* __restrict__ w_hh,
         const float* __restrict__ b_ih, const float* __restrict__ b_hh,
         float* __restrict__ gi, float* __restrict__ gh) {
  int g = blockIdx.x * blockDim.x + threadIdx.x;
  if (g >= 3 * H) return;
  float ai[BB], ah[BB];
#pragma unroll
  for (int b = 0; b < BB; ++b) { ai[b] = 0.f; ah[b] = 0.f; }
  const float* wi = w_ih + (size_t)g * (3 * H);
  for (int k = 0; k < 3 * H; ++k) {
    float wv = wi[k];
#pragma unroll
    for (int b = 0; b < BB; ++b) ai[b] += x[b * 3 * H + k] * wv;
  }
  const float* wh = w_hh + (size_t)g * H;
  for (int k = 0; k < H; ++k) {
    float wv = wh[k];
#pragma unroll
    for (int b = 0; b < BB; ++b) ah[b] += hid[b * H + k] * wv;
  }
  float bi = b_ih[g], bh = b_hh[g];
#pragma unroll
  for (int b = 0; b < BB; ++b) {
    gi[b * 3 * H + g] = ai[b] + bi;
    gh[b * 3 * H + g] = ah[b] + bh;
  }
}

// GRU gates (PyTorch order r,z,n), in-place hidden update.
__global__ void k_gru_gate(const float* __restrict__ gi, const float* __restrict__ gh,
                           float* __restrict__ hid) {
  int id = blockIdx.x * blockDim.x + threadIdx.x;
  if (id >= BB * H) return;
  int b = id / H, j = id % H;
  const float* gib = gi + b * 3 * H;
  const float* ghb = gh + b * 3 * H;
  float r = 1.f / (1.f + expf(-(gib[j] + ghb[j])));
  float z = 1.f / (1.f + expf(-(gib[H + j] + ghb[H + j])));
  float n = tanhf(gib[2 * H + j] + r * ghb[2 * H + j]);
  float h = hid[b * H + j];
  hid[b * H + j] = (1.f - z) * n + z * h;
}

// missing-mask precompute (token occurrence counts)
__global__ void k_cnt_zero(unsigned int* cnt) {
  int n = BB * VOC;
  for (int i = blockIdx.x * blockDim.x + threadIdx.x; i < n; i += gridDim.x * blockDim.x)
    cnt[i] = 0u;
}
__global__ void k_cnt_scatter(const int* __restrict__ inputs, unsigned int* cnt) {
  int id = blockIdx.x * blockDim.x + threadIdx.x;
  if (id >= BB * SEQ) return;
  int b = id / SEQ, s0 = id % SEQ;
  int tok = inputs[b * SEQ + s0];
  if (tok >= 0 && tok < VOC) atomicAdd(&cnt[(size_t)b * VOC + tok], 1u);
}

// init: SOS row of d_out, sampled[:,0]=1, hidden=sel=0, prev token = 1
__global__ void k_init(float* __restrict__ dout, float* hid, float* sel, int* smp,
                       const int* __restrict__ cls_to) {
  int id = blockIdx.x * blockDim.x + threadIdx.x;
  int n  = BB * EXT;
  for (int i = id; i < n; i += gridDim.x * blockDim.x) {
    int b = i / EXT, j = i % EXT;
    dout[(size_t)(b * MAXLEN) * EXT + j] = (j == 0) ? (float)(*cls_to) : 0.f;
  }
  if (id < BB * H) { hid[id] = 0.f; sel[id] = 0.f; }
  if (id < BB) {
    smp[id] = 1;
    dout[(size_t)BB * MAXLEN * EXT + id * MAXLEN] = 1.0f;
  }
}

// ---------------------------------------------------------------------------
// Combine: copy-score scatter, 61556-wide softmax, log-probs out, argmax,
// selective read for next step. One 1024-thread block per batch row.
// ---------------------------------------------------------------------------
__global__ void __launch_bounds__(1024)
k_combine(const float* __restrict__ gen, const float* __restrict__ css,
          const unsigned int* __restrict__ cnt, const int* __restrict__ inputs,
          const float* __restrict__ enc, float* __restrict__ cs_all,
          int* __restrict__ smp, float* __restrict__ sel,
          float* __restrict__ dout, int t) {
  __shared__ float red[1024];
  __shared__ int   redi[1024];
  __shared__ float ss[SEQ];
  const int b = blockIdx.x, tid = threadIdx.x, bs = blockDim.x;
  float* cs = cs_all + (size_t)b * EXT;
  const float* genb = gen + (size_t)b * GENP;

  for (int j = tid; j < EXT; j += bs)
    cs[j] = (j == 0 || j >= VOC || cnt[(size_t)b * VOC + j] == 0u) ? NEGV : 0.f;
  __syncthreads();
  for (int s0 = tid; s0 < SEQ; s0 += bs) {
    int tok = inputs[b * SEQ + s0];
    if (tok > 0 && tok < VOC) atomicAdd(&cs[tok], css[b * SEQ + s0]);
  }
  __syncthreads();

  float m = -3.4e38f;
  for (int j = tid; j < COMB; j += bs) {
    float v = (j < VOC) ? genb[j] : cs[j - VOC];
    m = fmaxf(m, v);
  }
  red[tid] = m; __syncthreads();
  for (int o = bs >> 1; o > 0; o >>= 1) {
    if (tid < o) red[tid] = fmaxf(red[tid], red[tid + o]);
    __syncthreads();
  }
  const float gmax = red[0];
  __syncthreads();

  float zs = 0.f;
  for (int j = tid; j < COMB; j += bs) {
    float v = (j < VOC) ? genb[j] : cs[j - VOC];
    zs += expf(v - gmax);
  }
  red[tid] = zs; __syncthreads();
  for (int o = bs >> 1; o > 0; o >>= 1) {
    if (tid < o) red[tid] += red[tid + o];
    __syncthreads();
  }
  const float rz = 1.f / red[0];
  __syncthreads();

  float* drow = dout + (size_t)(b * MAXLEN + t) * EXT;
  float bv = -1.f; int bi = 0;
  for (int j = tid; j < EXT; j += bs) {
    float p = expf(cs[j] - gmax) * rz;
    if (j < VOC) p += expf(genb[j] - gmax) * rz;
    drow[j] = logf(p + 1e-10f);
    if (p > bv) { bv = p; bi = j; }
  }
  red[tid] = bv; redi[tid] = bi; __syncthreads();
  for (int o = bs >> 1; o > 0; o >>= 1) {
    if (tid < o) {
      float v2 = red[tid + o]; int i2 = redi[tid + o];
      if (v2 > red[tid] || (v2 == red[tid] && i2 < redi[tid])) { red[tid] = v2; redi[tid] = i2; }
    }
    __syncthreads();
  }
  const int samp = redi[0];
  if (tid == 0) {
    smp[b] = samp;
    dout[(size_t)BB * MAXLEN * EXT + b * MAXLEN + t] = (float)samp;
  }
  __syncthreads();

  if (tid < SEQ) {
    int tok = inputs[b * SEQ + tid];
    ss[tid] = (tok == samp) ? css[b * SEQ + tid] : 0.f;
  }
  __syncthreads();
  red[tid] = (tid < SEQ) ? fabsf(ss[tid]) : 0.f; __syncthreads();
  for (int o = bs >> 1; o > 0; o >>= 1) {
    if (tid < o) red[tid] += red[tid + o];
    __syncthreads();
  }
  const float rden = 1.f / fmaxf(red[0], 1e-12f);
  __syncthreads();
  for (int j = tid; j < H; j += bs) {
    float a = 0.f;
    for (int s0 = 0; s0 < SEQ; ++s0) a += ss[s0] * enc[((size_t)b * SEQ + s0) * H + j];
    sel[b * H + j] = a * rden;
  }
}

// ---------------------------------------------------------------------------
extern "C" void kernel_launch(void* const* d_in, const int* in_sizes, int n_in,
                              void* d_out, int out_size, void* d_ws, size_t ws_size,
                              hipStream_t stream) {
  const float* enc    = (const float*)d_in[0];
  const int*   inputs = (const int*)d_in[1];
  const int*   cls_to = (const int*)d_in[2];
  /* d_in[3] sep_to unused by reference */
  const float* emb    = (const float*)d_in[4];
  const float* attn_w = (const float*)d_in[5];
  const float* attn_b = (const float*)d_in[6];
  const float* copy_w = (const float*)d_in[7];
  const float* copy_b = (const float*)d_in[8];
  const float* w_ih   = (const float*)d_in[9];
  const float* w_hh   = (const float*)d_in[10];
  const float* b_ih   = (const float*)d_in[11];
  const float* b_hh   = (const float*)d_in[12];
  const float* out_w  = (const float*)d_in[13];
  const float* out_b  = (const float*)d_in[14];
  float* dout = (float*)d_out;
  unsigned char* ws = (unsigned char*)d_ws;

  size_t off = 0;
  auto alloc = [&](size_t sz) { size_t o = off; off += (sz + 255) & ~(size_t)255; return o; };
  const size_t oB   = alloc((size_t)NT * KC * 32 * 32);   // bf16 B fragments (~44.7 MB)
  const size_t oA   = alloc((size_t)KC * 32 * 32);        // bf16 A fragments
  const size_t oH   = alloc((size_t)BB * H * 4);          // hidden
  const size_t oSel = alloc((size_t)BB * H * 4);          // selective read
  const size_t oCtx = alloc((size_t)BB * H * 4);          // context
  const size_t oX   = alloc((size_t)BB * 3 * H * 4);      // GRU input x
  const size_t oGi  = alloc((size_t)BB * 3 * H * 4);      // gi + b_ih
  const size_t oGh  = alloc((size_t)BB * 3 * H * 4);      // gh + b_hh
  const size_t oCss = alloc((size_t)BB * SEQ * 4);        // copy_score_seq
  const size_t oGen = alloc((size_t)BB * GENP * 4);       // gen scores (padded)
  const size_t oCs  = alloc((size_t)BB * EXT * 4);        // copy scores (ext vocab)
  const size_t oCnt = alloc((size_t)BB * VOC * 4);        // token occurrence counts
  const size_t oSmp = alloc((size_t)BB * 4);              // sampled tokens
  if (ws_size < off) return;  // workspace too small; nothing safe to do

  unsigned char* bpack = ws + oB;
  unsigned char* apack = ws + oA;
  float* hid = (float*)(ws + oH);
  float* sel = (float*)(ws + oSel);
  float* ctx = (float*)(ws + oCtx);
  float* xb  = (float*)(ws + oX);
  float* gi  = (float*)(ws + oGi);
  float* gh  = (float*)(ws + oGh);
  float* css = (float*)(ws + oCss);
  float* gen = (float*)(ws + oGen);
  float* cs  = (float*)(ws + oCs);
  unsigned int* cnt = (unsigned int*)(ws + oCnt);
  int* smp = (int*)(ws + oSmp);

  // One-time prep (per launch; deterministic)
  k_prep_bpack<<<NT * 3, 256, 0, stream>>>(out_w, bpack);           // NT*KC*32/256
  k_cnt_zero<<<512, 256, 0, stream>>>(cnt);
  k_cnt_scatter<<<(BB * SEQ + 255) / 256, 256, 0, stream>>>(inputs, cnt);
  k_init<<<(BB * EXT + 255) / 256, 256, 0, stream>>>(dout, hid, sel, smp, cls_to);

  for (int t = 1; t < MAXLEN; ++t) {
    k_attn<<<BB, 768, 0, stream>>>(enc, attn_w, attn_b, hid, ctx);
    k_build_x<<<(BB * 3 * H + 255) / 256, 256, 0, stream>>>(ctx, sel, emb, smp, xb);
    k_gru_mm<<<(3 * H + 255) / 256, 256, 0, stream>>>(xb, hid, w_ih, w_hh, b_ih, b_hh, gi, gh);
    k_gru_gate<<<(BB * H + 255) / 256, 256, 0, stream>>>(gi, gh, hid);
    k_copy<<<BB, 768, 0, stream>>>(enc, copy_w, copy_b, hid, css);
    k_pack_a<<<3, 256, 0, stream>>>(hid, apack);
    k_gen<<<(NT + 7) / 8, 256, 0, stream>>>(apack, bpack, out_b, gen);
    k_combine<<<BB, 1024, 0, stream>>>(gen, css, cnt, inputs, enc, cs, smp, sel, dout, t);
  }
}
